// LinearizedAttention_84456236909014
// MI455X (gfx1250) — matrix operand order repaired and checked by reading, not compiled, not run on hardware
//
#include <hip/hip_runtime.h>
#include <hip/hip_bf16.h>

typedef __attribute__((ext_vector_type(16))) _Float16 v16h;
typedef __attribute__((ext_vector_type(8)))  float    v8f;

#define WMMA(a, b, c) __builtin_amdgcn_wmma_f32_16x16x32_f16( \
    false, (a), false, (b), (short)0, (c), false, false)

#define BATCH 8
#define CHN   64
#define NPIX  65536
#define TILE  128
#define NTILE (NPIX / TILE)   // 512
#define BLKX  64              // blocks per batch -> 8 tiles per block

// workspace layout (bytes)
#define WS_KV    0            // 16384 f32 = 64 KB
#define WS_QKV16 65536        // 192*64 f16 = 24576 B (unpadded, for la_kv)
#define WS_WPAD  90112        // 2*64*66 f16 = 16896 B (wq rows 0..63 + wp, stride 66)

union U16 { v16h v; unsigned u[8]; };
union PK  { _Float16 h[2]; unsigned u; };

__device__ __forceinline__ float phi_f(float x) {
    return x > 0.0f ? x + 1.0f : __expf(x);   // elu(x)+1
}

__device__ __forceinline__ v8f vzero8() {
    v8f z;
#pragma unroll
    for (int i = 0; i < 8; ++i) z[i] = 0.0f;
    return z;
}

// stage a 64ch x 128px f32 tile as f16 transposed [t][c] (dword = c-pair)
__device__ __forceinline__ void stage_x(const float* __restrict__ xb, int base,
                                        unsigned* __restrict__ ldsx_dw, int tid) {
#pragma unroll
    for (int it = 0; it < 4; ++it) {
        const int i  = tid + it * 256;     // 0..1023
        const int c2 = i >> 5;             // channel pair 0..31
        const int t4 = (i & 31) << 2;      // pixel base 0..124
        const float4 a = *(const float4*)(xb + ((2 * c2) * NPIX + base + t4));
        const float4 b = *(const float4*)(xb + ((2 * c2 + 1) * NPIX + base + t4));
        PK p;
        p.h[0] = (_Float16)a.x; p.h[1] = (_Float16)b.x; ldsx_dw[(t4 + 0) * 33 + c2] = p.u;
        p.h[0] = (_Float16)a.y; p.h[1] = (_Float16)b.y; ldsx_dw[(t4 + 1) * 33 + c2] = p.u;
        p.h[0] = (_Float16)a.z; p.h[1] = (_Float16)b.z; ldsx_dw[(t4 + 2) * 33 + c2] = p.u;
        p.h[0] = (_Float16)a.w; p.h[1] = (_Float16)b.w; ldsx_dw[(t4 + 3) * 33 + c2] = p.u;
    }
}

// ---------------------------------------------------------------------------
// Kernel 0: zero KV, convert weights f32->f16 (flat + LDS-stride-padded copies)
// ---------------------------------------------------------------------------
__global__ __launch_bounds__(256) void la_prep(
    const float* __restrict__ qkv_w, const float* __restrict__ proj_w,
    float* __restrict__ KV, _Float16* __restrict__ wq16,
    _Float16* __restrict__ wpad) {
    int idx = blockIdx.x * 256 + threadIdx.x;          // 0..16383
    KV[idx] = 0.0f;                                    // 8*2*32*32 = 16384
    if (idx < 192 * 64) wq16[idx] = (_Float16)qkv_w[idx];
    if (idx < 64 * 64) {                               // padded copies, stride 66
        int r = idx >> 6, c = idx & 63;
        wpad[r * 66 + c]           = (_Float16)qkv_w[idx];   // q rows 0..63
        wpad[64 * 66 + r * 66 + c] = (_Float16)proj_w[idx];  // proj
    }
}

// ---------------------------------------------------------------------------
// Kernel 1: k = Wk x, v = Wv x, KV[b,h] += phi(k) @ v^T  (reduction over n)
// ---------------------------------------------------------------------------
__global__ __launch_bounds__(256) void la_kv(
    const float* __restrict__ x, const _Float16* __restrict__ wq16,
    float* __restrict__ KV) {
    __shared__ alignas(16) _Float16 lds_x[TILE * 66];    // [t][c], stride 66
    __shared__ alignas(16) _Float16 lds_kv[8][128 * 18]; // per wave, [row][t]

    const int tid   = threadIdx.x;
    const int lane  = tid & 31;
    const int wv    = tid >> 5;
    const int lrow  = lane & 15;
    const bool hi   = lane >= 16;
    const int batch = blockIdx.y;
    const float* xb = x + (size_t)batch * CHN * NPIX;
    const unsigned* gw = (const unsigned*)wq16;          // dword view, 32 dw/row

    v8f accKV[2][2][2];
#pragma unroll
    for (int h = 0; h < 2; ++h)
#pragma unroll
        for (int mt = 0; mt < 2; ++mt)
#pragma unroll
            for (int nt = 0; nt < 2; ++nt) accKV[h][mt][nt] = vzero8();

    const int t0 = wv * 16;
    const int abase = hi ? 8 : 0;

    for (int tile = blockIdx.x; tile < NTILE; tile += gridDim.x) {
        const int base = tile * TILE;
        __syncthreads();
        stage_x(xb, base, (unsigned*)lds_x, tid);
        int ntile = tile + gridDim.x;
        if (ntile < NTILE)  // prefetch one 128B line of next tile per thread
            __builtin_prefetch(xb + (tid >> 2) * NPIX + ntile * TILE +
                                   (tid & 3) * 32, 0, 0);
        __syncthreads();

        // B operand: x[c 0..31][t], x[c 32..63][t] for this wave's 16 px
        U16 bx0, bx1;
        {
            const unsigned* px = (const unsigned*)&lds_x[(t0 + lrow) * 66];
            const int kb = hi ? 8 : 0;
#pragma unroll
            for (int j = 0; j < 8; ++j) {
                bx0.u[j] = px[kb + j];
                bx1.u[j] = px[16 + kb + j];
            }
        }

        _Float16* skv = lds_kv[wv];
        // m = 0..3 -> k (qkv rows 64..127), m = 4..7 -> v (rows 128..191)
#pragma unroll
        for (int m = 0; m < 8; ++m) {
            U16 a0, a1;
            const int rowdw = (64 + m * 16 + lrow) * 32;
#pragma unroll
            for (int j = 0; j < 8; ++j) {
                const int jj = j;                        // K pairs contiguous {0,2,..,14}
                a0.u[j] = gw[rowdw + abase + jj];
                a1.u[j] = gw[rowdw + 16 + abase + jj];
            }
            v8f acc = vzero8();
            acc = WMMA(a0.v, bx0.v, acc);
            acc = WMMA(a1.v, bx1.v, acc);
#pragma unroll
            for (int r = 0; r < 8; ++r) {
                float val = acc[r];
                if (m < 4) val = phi_f(val);             // phi on k only
                skv[(m * 16 + r + (hi ? 8 : 0)) * 18 + lrow] = (_Float16)val;
            }
        }

        // KV[h] += k_phi (A, K=t zero-padded to 32) x v^T (B)
#pragma unroll
        for (int h = 0; h < 2; ++h) {
#pragma unroll
            for (int mt = 0; mt < 2; ++mt) {
                U16 a;
                const unsigned* pk =
                    (const unsigned*)&skv[(h * 32 + mt * 16 + lrow) * 18];
#pragma unroll
                for (int j = 0; j < 8; ++j)
                    a.u[j] = hi ? 0u : pk[j];
#pragma unroll
                for (int nt = 0; nt < 2; ++nt) {
                    U16 bV;
                    const unsigned* pv =
                        (const unsigned*)&skv[(64 + h * 32 + nt * 16 + lrow) * 18];
#pragma unroll
                    for (int j = 0; j < 8; ++j)
                        bV.u[j] = hi ? 0u : pv[j];
                    accKV[h][mt][nt] = WMMA(a.v, bV.v, accKV[h][mt][nt]);
                }
            }
        }
    }

    float* kvb = KV + batch * 2048;
#pragma unroll
    for (int h = 0; h < 2; ++h)
#pragma unroll
        for (int mt = 0; mt < 2; ++mt)
#pragma unroll
            for (int nt = 0; nt < 2; ++nt)
#pragma unroll
                for (int r = 0; r < 8; ++r) {
                    const int c = mt * 16 + r + (hi ? 8 : 0);
                    const int d = nt * 16 + lrow;
                    atomicAdd(&kvb[h * 1024 + c * 32 + d], accKV[h][mt][nt][r]);
                }
}

// ---------------------------------------------------------------------------
// Kernel 2: q = Wq x; out = KV^T @ phi(q) per head; y = Wp @ out
// ---------------------------------------------------------------------------
__global__ __launch_bounds__(256) void la_out(
    const float* __restrict__ x, const _Float16* __restrict__ wpad,
    const float* __restrict__ KV, float* __restrict__ y) {
    __shared__ alignas(16) _Float16 lds_w[2 * 64 * 66]; // wq rows 0..63 | wp
    __shared__ alignas(16) _Float16 lds_kvT[2 * 32 * 34];
    __shared__ alignas(16) _Float16 lds_x[TILE * 66];
    __shared__ alignas(16) _Float16 lds_s[8][16 * 66];

    const int tid   = threadIdx.x;
    const int lane  = tid & 31;
    const int wv    = tid >> 5;
    const int lrow  = lane & 15;
    const bool hi   = lane >= 16;
    const int batch = blockIdx.y;
    const float* xb = x + (size_t)batch * CHN * NPIX;

    // stage padded weights via async global->LDS copy (16896 B = 1056 x 16 B)
    {
        const unsigned lds_base = (unsigned)(size_t)(void*)&lds_w[0];
        const char* gsrc = (const char*)wpad;
        for (int i = tid; i < 1056; i += 256) {
            unsigned la = lds_base + i * 16;
            const char* ga = gsrc + i * 16;
            asm volatile("global_load_async_to_lds_b128 %0, %1, off"
                         :: "v"(la), "v"(ga) : "memory");
        }
        asm volatile("s_wait_asynccnt 0x0" ::: "memory");
    }
    // stage KV transposed to [h][d][c] f16 (A-layout friendly)
    const float* kvb = KV + batch * 2048;
    for (int i = tid; i < 2048; i += 256) {
        int h = i >> 10, rem = i & 1023, c = rem >> 5, d = rem & 31;
        lds_kvT[(h * 32 + d) * 34 + c] = (_Float16)kvb[h * 1024 + c * 32 + d];
    }

    const _Float16* lds_wq = lds_w;
    const _Float16* lds_wp = lds_w + 64 * 66;
    const int t0 = wv * 16;
    const int abase = hi ? 8 : 0;

    for (int tile = blockIdx.x; tile < NTILE; tile += gridDim.x) {
        const int base = tile * TILE;
        __syncthreads();
        stage_x(xb, base, (unsigned*)lds_x, tid);
        int ntile = tile + gridDim.x;
        if (ntile < NTILE)
            __builtin_prefetch(xb + (tid >> 2) * NPIX + ntile * TILE +
                                   (tid & 3) * 32, 0, 0);
        __syncthreads();

        U16 bx0, bx1;
        {
            const unsigned* px = (const unsigned*)&lds_x[(t0 + lrow) * 66];
            const int kb = hi ? 8 : 0;
#pragma unroll
            for (int j = 0; j < 8; ++j) {
                bx0.u[j] = px[kb + j];
                bx1.u[j] = px[16 + kb + j];
            }
        }

        _Float16* ss = lds_s[wv];
        // q = Wq x ; phi ; stage as [t][c]
#pragma unroll
        for (int m = 0; m < 4; ++m) {
            U16 a0, a1;
            const unsigned* pw = (const unsigned*)&lds_wq[(m * 16 + lrow) * 66];
#pragma unroll
            for (int j = 0; j < 8; ++j) {
                const int jj = j;
                a0.u[j] = pw[abase + jj];
                a1.u[j] = pw[16 + abase + jj];
            }
            v8f acc = vzero8();
            acc = WMMA(a0.v, bx0.v, acc);
            acc = WMMA(a1.v, bx1.v, acc);
#pragma unroll
            for (int r = 0; r < 8; ++r) {
                const int c = m * 16 + r + (hi ? 8 : 0);
                ss[lrow * 66 + c] = (_Float16)phi_f(acc[r]);
            }
        }

        // out[h] = KV^T @ q_phi : A = KV^T [d x c], B = q_phi [c x t]
        v8f accO[2][2];
#pragma unroll
        for (int h = 0; h < 2; ++h) {
            const unsigned* pq = (const unsigned*)&ss[lrow * 66];
#pragma unroll
            for (int dt = 0; dt < 2; ++dt) {
                U16 a;
                const unsigned* pk =
                    (const unsigned*)&lds_kvT[(h * 32 + dt * 16 + lrow) * 34];
#pragma unroll
                for (int j = 0; j < 8; ++j) {
                    const int jj = j;
                    a.u[j] = pk[abase + jj];
                }
                U16 bq;
#pragma unroll
                for (int j = 0; j < 8; ++j)
                    bq.u[j] = pq[h * 16 + (hi ? 8 : 0) + j];
                accO[h][dt] = WMMA(a.v, bq.v, vzero8());
            }
        }
        // re-stage out as [t][c] (same-wave DS ops are in order)
#pragma unroll
        for (int h = 0; h < 2; ++h)
#pragma unroll
            for (int dt = 0; dt < 2; ++dt)
#pragma unroll
                for (int r = 0; r < 8; ++r) {
                    const int co = h * 32 + dt * 16 + r + (hi ? 8 : 0);
                    ss[lrow * 66 + co] = (_Float16)accO[h][dt][r];
                }

        // y = Wp @ out
#pragma unroll
        for (int m = 0; m < 4; ++m) {
            U16 a0, a1;
            const unsigned* pw = (const unsigned*)&lds_wp[(m * 16 + lrow) * 66];
#pragma unroll
            for (int j = 0; j < 8; ++j) {
                const int jj = j;
                a0.u[j] = pw[abase + jj];
                a1.u[j] = pw[16 + abase + jj];
            }
            U16 b0, b1;
            const unsigned* po = (const unsigned*)&ss[lrow * 66];
#pragma unroll
            for (int j = 0; j < 8; ++j) {
                b0.u[j] = po[(hi ? 8 : 0) + j];
                b1.u[j] = po[16 + (hi ? 8 : 0) + j];
            }
            v8f acc = vzero8();
            acc = WMMA(a0.v, b0.v, acc);
            acc = WMMA(a1.v, b1.v, acc);
#pragma unroll
            for (int r = 0; r < 8; ++r) {
                const int c = m * 16 + r + (hi ? 8 : 0);
                y[(batch * CHN + c) * NPIX + base + t0 + lrow] = acc[r];
            }
        }
    }
}

extern "C" void kernel_launch(void* const* d_in, const int* in_sizes, int n_in,
                              void* d_out, int out_size, void* d_ws,
                              size_t ws_size, hipStream_t stream) {
    const float* x      = (const float*)d_in[0];
    const float* qkv_w  = (const float*)d_in[1];
    const float* proj_w = (const float*)d_in[2];
    float*     KV   = (float*)((char*)d_ws + WS_KV);
    _Float16*  wq16 = (_Float16*)((char*)d_ws + WS_QKV16);
    _Float16*  wpad = (_Float16*)((char*)d_ws + WS_WPAD);
    float* y = (float*)d_out;

    la_prep<<<64, 256, 0, stream>>>(qkv_w, proj_w, KV, wq16, wpad);
    la_kv<<<dim3(BLKX, BATCH), 256, 0, stream>>>(x, wq16, KV);
    la_out<<<dim3(BLKX, BATCH), 256, 0, stream>>>(x, wpad, KV, y);
}
